// NASRNN_55061480735404
// MI455X (gfx1250) — matrix operand order repaired and checked by reading, not compiled, not run on hardware
//
#include <hip/hip_runtime.h>
#include <hip/hip_bf16.h>

#define HID  256
#define CIN  128
#define BSZ  128
#define TT   2048
#define NG   1024          // 4*HID
#define NBLK 16            // persistent workgroups (16 hidden units each)
#define NTHR 256           // 8 wave32
#define XP   136           // LDS pitch for x tile (ushorts), 16B aligned rows
#define GP   68            // LDS pitch for gates tile (floats)
#define NKT  12            // K tiles: 4 (x, K=128) + 8 (h, K=256)

typedef __attribute__((ext_vector_type(16))) __bf16 v16bf;
typedef __attribute__((ext_vector_type(8)))  float  v8f;

union FragB16 { v16bf v; uint4 q[2]; };
union FragF32 { v8f  v; float f[8]; };

// workspace layout (bytes)
#define WS_WIN   0                           // 1024x128 bf16
#define WS_WHID  (WS_WIN  + NG*CIN*2)        // 1024x256 bf16
#define WS_H0    (WS_WHID + NG*HID*2)        // 128x256 bf16 (ping)
#define WS_H1    (WS_H0   + BSZ*HID*2)       // 128x256 bf16 (pong)
#define WS_GW    (WS_H1   + BSZ*HID*2)       // 256 f32
#define WS_BAR   (WS_GW   + HID*4)           // 2 u32 (counter, generation)

static __device__ __forceinline__ unsigned short f2bf(float f) {
    unsigned int u = __float_as_uint(f);
    unsigned int r = (u + 0x7FFFu + ((u >> 16) & 1u)) >> 16;   // RNE
    return (unsigned short)r;
}

__global__ void nasrnn_prep(const float* __restrict__ w_in,
                            const float* __restrict__ w_hid,
                            const float* __restrict__ alphas,
                            unsigned short* __restrict__ wbf_in,
                            unsigned short* __restrict__ wbf_hid,
                            unsigned short* __restrict__ h0,
                            unsigned short* __restrict__ h1,
                            float* __restrict__ gw,
                            unsigned* __restrict__ bar,
                            float* __restrict__ out_scalars)
{
    int tid = blockIdx.x * blockDim.x + threadIdx.x;
    int stride = gridDim.x * blockDim.x;
    for (int i = tid; i < NG * CIN; i += stride) wbf_in[i]  = f2bf(w_in[i]);
    for (int i = tid; i < NG * HID; i += stride) wbf_hid[i] = f2bf(w_hid[i]);
    for (int i = tid; i < BSZ * HID; i += stride) { h0[i] = 0; h1[i] = 0; }

    float a0 = alphas[0], a1 = alphas[1], a2 = alphas[2];
    float m  = fmaxf(a0, fmaxf(a1, a2));
    float e0 = __expf(a0 - m), e1 = __expf(a1 - m), e2 = __expf(a2 - m);
    float s  = e0 + e1 + e2;
    float p0 = e0 / s, p1 = e1 / s, p2 = e2 / s;

    for (int j = tid; j < HID; j += stride) {
        float g = p2 + ((j < 153) ? p1 : 0.f) + ((j < 51) ? p0 : 0.f);
        gw[j] = g;
    }
    if (tid == 0) {
        bar[0] = 0u; bar[1] = 0u;
        float expected = 51.f * p0 + 153.f * p1 + 256.f * p2;
        out_scalars[0] = expected;                                  // expected channels
        out_scalars[1] = expected * (8.f + (float)TT * 13.f * 4.f); // model+act size
    }
}

__global__ void __launch_bounds__(NTHR, 1)
nasrnn_fused(const float* __restrict__ x,
             const unsigned short* __restrict__ wbf_in,
             const unsigned short* __restrict__ wbf_hid,
             unsigned short* __restrict__ hbuf0,
             unsigned short* __restrict__ hbuf1,
             const float* __restrict__ gw,
             unsigned* __restrict__ bar,
             float* __restrict__ hout)
{
    __shared__ unsigned short xs[BSZ * XP];   // x_t tile, bf16          (34,816 B)
    __shared__ float gates[BSZ * GP];         // 128x64 gate tile, f32   (34,816 B)
    __shared__ float cstate[BSZ * 16];        // persistent cell state   ( 8,192 B)
    __shared__ float gws[16];

    const int tid   = threadIdx.x;
    const int lane  = tid & 31;
    const int wave  = tid >> 5;               // 0..7 -> M tile (16 batch rows)
    const int j0    = blockIdx.x * 16;        // hidden-unit slice for this WG

    const int ncol   = lane & 15;             // B/D column within tile
    const int khalfA = (lane >> 4) * 8;       // A frag K sub-offset (16-bit layout)
    const int khalfB = (lane >> 4) * 8;       // B frag K sub-offset
    const int mrowA  = wave * 16 + (lane & 15);
    const int mbaseD = wave * 16 + ((lane >> 4) << 3);

    if (tid < 16) gws[tid] = gw[j0 + tid];
    for (int i = tid; i < BSZ * 16; i += NTHR) cstate[i] = 0.f;

    volatile unsigned* vbar = bar;

    // Per-lane A-fragment base addresses (K-tile advances by 32 bf16 = 64 B)
    const unsigned short* const xsBase = &xs[mrowA * XP + khalfA];

    for (int t = 0; t < TT; ++t) {
        const unsigned short* hr = (t & 1) ? hbuf1 : hbuf0;
        unsigned short*       hw = (t & 1) ? hbuf0 : hbuf1;
        const unsigned short* hBase = hr + mrowA * HID + khalfA;

        // ---- stage x[:, :, t] into LDS as bf16 (x is (B, CIN, T)) ----
        for (int i = tid; i < BSZ * CIN; i += NTHR) {
            int b = i >> 7, c = i & 127;
            const float* xp = &x[((b << 7) + c) * TT + t];
            xs[b * XP + c] = f2bf(*xp);
            if (t + 1 < TT) __builtin_prefetch(xp + 1, 0, 0);   // global_prefetch_b8
        }
        __syncthreads();

        // ---- WMMA: gates(128x64) = [x_t | h](128x384) * W^T slice (384x64) ----
        // B fragments (weights) are loop-invariant -> compiler keeps them in VGPRs.
        // A fragments are software-pipelined (double-buffered) so each pair of
        // ds/global b128 loads overlaps the previous 4 WMMAs.
        v8f acc[4] = {};
        FragB16 a[2];

        // preload K-tile 0 (x part, from LDS)
        a[0].q[0] = *(const uint4*)(xsBase);
        a[0].q[1] = *(const uint4*)(xsBase + 16);

        #pragma unroll
        for (int kt = 0; kt < NKT; ++kt) {
            // issue loads for next K-tile before consuming current one
            if (kt + 1 < NKT) {
                FragB16& an = a[(kt + 1) & 1];
                if (kt + 1 < 4) {
                    const unsigned short* p = xsBase + (kt + 1) * 32;
                    an.q[0] = *(const uint4*)p;
                    an.q[1] = *(const uint4*)(p + 16);
                } else {
                    const unsigned short* p = hBase + (kt + 1 - 4) * 32;
                    an.q[0] = *(const uint4*)p;
                    an.q[1] = *(const uint4*)(p + 16);
                }
            }
            const FragB16& ac = a[kt & 1];
            const int kb = (kt < 4) ? kt * 32 : (kt - 4) * 32;
            const unsigned short* wbase =
                (kt < 4) ? (wbf_in  + (j0 + ncol) * CIN + kb + khalfB)
                         : (wbf_hid + (j0 + ncol) * HID + kb + khalfB);
            const int wpitch = (kt < 4) ? (HID * CIN) : (HID * HID);
            #pragma unroll
            for (int nt = 0; nt < 4; ++nt) {
                FragB16 bf;
                const unsigned short* wp = wbase + nt * wpitch;
                bf.q[0] = *(const uint4*)wp;
                bf.q[1] = *(const uint4*)(wp + 16);
                acc[nt] = __builtin_amdgcn_wmma_f32_16x16x32_bf16(
                    false, ac.v, false, bf.v, (short)0, acc[nt], false, false);
            }
        }

        // ---- scatter accumulators to LDS gates tile (C/D layout) ----
        #pragma unroll
        for (int nt = 0; nt < 4; ++nt) {
            FragF32 fr; fr.v = acc[nt];
            #pragma unroll
            for (int r = 0; r < 8; ++r)
                gates[(mbaseD + r) * GP + nt * 16 + ncol] = fr.f[r];
        }
        __syncthreads();

        // ---- masked LSTM elementwise update (128 x 16 hidden units) ----
        #pragma unroll
        for (int i = 0; i < 8; ++i) {
            int w  = tid + i * NTHR;           // 0..2047
            int b  = w >> 4, jl = w & 15;
            float g  = gws[jl];
            float ig = gates[b * GP + jl];
            float fg = gates[b * GP + 16 + jl];
            float cg = gates[b * GP + 32 + jl];
            float og = gates[b * GP + 48 + jl];
            ig = 1.f / (1.f + __expf(-ig * g));
            fg = 1.f / (1.f + __expf(-fg * g));
            og = 1.f / (1.f + __expf(-og * g));
            float c = fg * cstate[b * 16 + jl] + ig * tanhf(cg * g);
            cstate[b * 16 + jl] = c;
            float h = og * tanhf(c);
            hw[b * HID + j0 + jl] = f2bf(h);
            if (t == TT - 1) hout[b * HID + j0 + jl] = h;
        }
        __threadfence();

        // ---- grid-wide sense-reversing barrier (16 co-resident blocks) ----
        __syncthreads();
        if (tid == 0) {
            unsigned g = vbar[1];
            if (atomicAdd(&bar[0], 1u) == NBLK - 1) {
                bar[0] = 0u;
                __threadfence();
                vbar[1] = g + 1u;
            } else {
                while (vbar[1] == g) __builtin_amdgcn_s_sleep(1);
            }
        }
        __syncthreads();
    }
}

extern "C" void kernel_launch(void* const* d_in, const int* in_sizes, int n_in,
                              void* d_out, int out_size, void* d_ws, size_t ws_size,
                              hipStream_t stream) {
    const float* x      = (const float*)d_in[0];   // (128, 128, 2048)
    const float* w_in   = (const float*)d_in[1];   // (1024, 128)
    const float* w_hid  = (const float*)d_in[2];   // (1024, 256)
    const float* alphas = (const float*)d_in[3];   // (3,)
    (void)in_sizes; (void)n_in; (void)out_size; (void)ws_size;

    char* ws = (char*)d_ws;
    unsigned short* wbf_in  = (unsigned short*)(ws + WS_WIN);
    unsigned short* wbf_hid = (unsigned short*)(ws + WS_WHID);
    unsigned short* h0      = (unsigned short*)(ws + WS_H0);
    unsigned short* h1      = (unsigned short*)(ws + WS_H1);
    float*          gw      = (float*)(ws + WS_GW);
    unsigned*       bar     = (unsigned*)(ws + WS_BAR);

    float* hout        = (float*)d_out;            // 128x256 fp32
    float* out_scalars = hout + BSZ * HID;         // expected, sizes

    nasrnn_prep<<<64, NTHR, 0, stream>>>(w_in, w_hid, alphas,
                                         wbf_in, wbf_hid, h0, h1,
                                         gw, bar, out_scalars);
    nasrnn_fused<<<NBLK, NTHR, 0, stream>>>(x, wbf_in, wbf_hid, h0, h1,
                                            gw, bar, hout);
}
